// WaveNet_58746562674767
// MI455X (gfx1250) — matrix operand order repaired and checked by reading, not compiled, not run on hardware
//
#include <hip/hip_runtime.h>
#include <math.h>

// ---------------- problem constants ----------------
#define RES    256
#define SKIP   512
#define NCLS   256
#define KTAPS  2
#define NL     30
#define BATCH  4
#define TLEN   8192
#define WWIN   4096     // trailing window (receptive field 3070 <= 4096), 64 tiles
#define TN     64       // positions per tile

typedef __attribute__((ext_vector_type(16))) _Float16 v16h;
typedef __attribute__((ext_vector_type(8)))  _Float16 v8h;
typedef __attribute__((ext_vector_type(8)))  float    v8f;

#define HS_STRIDE 520   // halves per LDS row for staged H (512 + 8 pad -> bank stride 4)
#define GS_STRIDE 264   // halves per LDS row for gated   (256 + 8 pad)
#define SMEM_BYTES (TN*HS_STRIDE*2 + TN*GS_STRIDE*2)   // 66560 + 33792 = 100352 B

// =====================================================================
// Weight repack: f32 -> f16 in exact WMMA A-fragment order.
// A-fragment (16x32 f16, ISA 7.12.2): lane L holds row m = L%16, half = L/16;
// half j (vgpr v=j/2, pos=j%2) has koff = 8*half + 2*(v&3) + 16*(v>>2) + pos.
// Per-layer layout: [kt][mtile][lane][j] -> lane reads 32 contiguous bytes.
// =====================================================================
__global__ void repack_wd(const float* __restrict__ Wd, _Float16* __restrict__ WdP) {
    int idx = blockIdx.x * 256 + threadIdx.x;
    if (idx >= NL * 16 * 16 * 512) return;
    int j     = idx & 15;
    int lane  = (idx >> 4) & 31;
    int mt    = (idx >> 9) & 15;
    int kt    = (idx >> 13) & 15;
    int layer = idx >> 17;
    int m  = mt * 16 + (lane & 15);
    int hg = lane >> 4;
    int v  = j >> 1, pos = j & 1;
    int koff = 8 * hg + 2 * (v & 3) + 16 * (v >> 2) + pos;
    int k    = kt * 32 + koff;        // 0..511: k<256 -> tap0 (h[t-d]), else tap1 (h[t])
    int tap  = k >> 8;
    int ci   = k & (RES - 1);
    WdP[idx] = (_Float16)Wd[(((size_t)layer * RES + m) * RES + ci) * KTAPS + tap];
}

__global__ void repack_wr(const float* __restrict__ Wr, _Float16* __restrict__ WrP) {
    int idx = blockIdx.x * 256 + threadIdx.x;
    if (idx >= NL * 8 * 16 * 512) return;
    int j     = idx & 15;
    int lane  = (idx >> 4) & 31;
    int mt    = (idx >> 9) & 15;
    int kt    = (idx >> 13) & 7;
    int layer = idx >> 16;
    int m  = mt * 16 + (lane & 15);
    int hg = lane >> 4;
    int v  = j >> 1, pos = j & 1;
    int koff = 8 * hg + 2 * (v & 3) + 16 * (v >> 2) + pos;
    int ci   = kt * 32 + koff;        // 0..255
    WrP[idx] = (_Float16)Wr[((size_t)layer * RES + m) * RES + ci];
}

// =====================================================================
// Start conv (1 -> 256 channels, K=2, d=1) on the trailing window.
// h stored position-major: h[b][w][co] so channel vectors are contiguous.
// =====================================================================
__global__ void start_conv(const float* __restrict__ x, const float* __restrict__ Wst,
                           const float* __restrict__ bst, float* __restrict__ h0) {
    int w = blockIdx.x, b = blockIdx.y, co = threadIdx.x;
    int t = TLEN - WWIN + w;                 // t >= 4096, t-1 >= 4095 in bounds
    float xl = x[(size_t)b * TLEN + t - 1];
    float xc = x[(size_t)b * TLEN + t];
    h0[((size_t)b * WWIN + w) * RES + co] = Wst[co * 2] * xl + Wst[co * 2 + 1] * xc + bst[co];
}

// =====================================================================
// Fused WaveNet layer. One block = 256x64 tile (8 wave32 waves).
//   phase0: stage [h(t-d); h(t)] (512 x 64) f32->f16 into LDS
//   phase1: pre = Wd @ H            (WMMA, K=512)
//   phase2: gated = tanh*sigmoid(pre+bd) -> f16 LDS; tile 0 exports t=T-1 col
//   phase3: h_new = h_old + Wr @ gated + br   (WMMA, K=256)
// =====================================================================
__global__ __launch_bounds__(256) void wavenet_layer(
    const float* __restrict__ hin, float* __restrict__ hout,
    const _Float16* __restrict__ wdp, const _Float16* __restrict__ wrp,
    const float* __restrict__ bd, const float* __restrict__ br,
    float* __restrict__ glast, int dil)
{
    extern __shared__ char smem[];
    _Float16* Hs = (_Float16*)smem;                        // [TN][HS_STRIDE]
    _Float16* Gs = (_Float16*)(smem + TN * HS_STRIDE * 2); // [TN][GS_STRIDE]

    const int tid  = threadIdx.x;
    const int b    = blockIdx.y;
    const int w0   = WWIN - TN * (blockIdx.x + 1);         // tile 0 = last positions
    const int wave = tid >> 5;
    const int lane = tid & 31;
    const int lid  = lane & 15;
    const int hg   = lane >> 4;
    const int m0   = wave * 32;

    // ---------------- phase 0: stage H into LDS ----------------
    {
        int n    = tid >> 2;            // 0..63 row
        int part = tid & 3;             // 128 halves each
        int kb   = part * 128;
        int wsrc = (kb < RES) ? (w0 + n - dil) : (w0 + n);
        if (wsrc < 0) wsrc = 0;         // never triggers by construction
        int cib  = kb & (RES - 1);
        const float4* src = (const float4*)(hin + ((size_t)b * WWIN + wsrc) * RES + cib);
        _Float16* dst = Hs + n * HS_STRIDE + kb;
        #pragma unroll 8
        for (int q = 0; q < 32; ++q) {
            float4 vv = src[q];
            union { _Float16 h[4]; unsigned long long u; } pk;
            pk.h[0] = (_Float16)vv.x; pk.h[1] = (_Float16)vv.y;
            pk.h[2] = (_Float16)vv.z; pk.h[3] = (_Float16)vv.w;
            *(unsigned long long*)(dst + q * 4) = pk.u;
        }
    }
    __syncthreads();

    v8f vzero = {};
    // ---------------- phase 1: pre = Wd @ [h(t-d); h(t)] ----------------
    v8f acc[2][4];
    #pragma unroll
    for (int mt = 0; mt < 2; ++mt)
        #pragma unroll
        for (int nt = 0; nt < 4; ++nt) acc[mt][nt] = vzero;

    for (int kt = 0; kt < 16; ++kt) {
        v16h a[2];
        #pragma unroll
        for (int mt = 0; mt < 2; ++mt)
            a[mt] = *(const v16h*)(wdp + ((size_t)(kt * 16 + wave * 2 + mt)) * 512 + lane * 16);
        #pragma unroll
        for (int nt = 0; nt < 4; ++nt) {
            const _Float16* p = Hs + (nt * 16 + lid) * HS_STRIDE + kt * 32 + hg * 8;
            v8h blo = *(const v8h*)p;
            v8h bhi = *(const v8h*)(p + 16);
            v16h bf = __builtin_shufflevector(blo, bhi, 0,1,2,3,4,5,6,7,8,9,10,11,12,13,14,15);
            #pragma unroll
            for (int mt = 0; mt < 2; ++mt)
                acc[mt][nt] = __builtin_amdgcn_wmma_f32_16x16x32_f16(
                    false, a[mt], false, bf, (short)0, acc[mt][nt], false, false);
        }
    }

    // ---------------- phase 2: gating -> Gs ----------------
    #pragma unroll
    for (int mt = 0; mt < 2; ++mt) {
        const float4* bp = (const float4*)(bd + m0 + mt * 16 + hg * 8);
        float4 b0 = bp[0], b1 = bp[1];
        float bdv[8] = { b0.x, b0.y, b0.z, b0.w, b1.x, b1.y, b1.z, b1.w };
        #pragma unroll
        for (int nt = 0; nt < 4; ++nt) {
            int n = nt * 16 + lid;
            union { _Float16 h[8]; uint4 u; } pk;
            #pragma unroll
            for (int r = 0; r < 8; ++r) {
                float xv = acc[mt][nt][r] + bdv[r];
                float g  = tanhf(xv) * (1.0f / (1.0f + __expf(-xv)));
                pk.h[r] = (_Float16)g;
            }
            *(uint4*)(Gs + n * GS_STRIDE + m0 + mt * 16 + hg * 8) = pk.u;
        }
    }
    __syncthreads();

    // tile containing t = T-1 exports gated column for the skip head
    if (blockIdx.x == 0)
        glast[(size_t)b * RES + tid] = (float)Gs[63 * GS_STRIDE + tid];

    // ---------------- phase 3: h_new = h_old + Wr @ gated + br ----------------
    v8f acc2[2][4];
    #pragma unroll
    for (int mt = 0; mt < 2; ++mt)
        #pragma unroll
        for (int nt = 0; nt < 4; ++nt) acc2[mt][nt] = vzero;

    for (int kt = 0; kt < 8; ++kt) {
        v16h a[2];
        #pragma unroll
        for (int mt = 0; mt < 2; ++mt)
            a[mt] = *(const v16h*)(wrp + ((size_t)(kt * 16 + wave * 2 + mt)) * 512 + lane * 16);
        #pragma unroll
        for (int nt = 0; nt < 4; ++nt) {
            const _Float16* p = Gs + (nt * 16 + lid) * GS_STRIDE + kt * 32 + hg * 8;
            v8h blo = *(const v8h*)p;
            v8h bhi = *(const v8h*)(p + 16);
            v16h bf = __builtin_shufflevector(blo, bhi, 0,1,2,3,4,5,6,7,8,9,10,11,12,13,14,15);
            #pragma unroll
            for (int mt = 0; mt < 2; ++mt)
                acc2[mt][nt] = __builtin_amdgcn_wmma_f32_16x16x32_f16(
                    false, a[mt], false, bf, (short)0, acc2[mt][nt], false, false);
        }
    }

    #pragma unroll
    for (int mt = 0; mt < 2; ++mt) {
        const float4* bp = (const float4*)(br + m0 + mt * 16 + hg * 8);
        float4 b0 = bp[0], b1 = bp[1];
        float brv[8] = { b0.x, b0.y, b0.z, b0.w, b1.x, b1.y, b1.z, b1.w };
        #pragma unroll
        for (int nt = 0; nt < 4; ++nt) {
            int n = nt * 16 + lid;
            size_t base = ((size_t)b * WWIN + (size_t)(w0 + n)) * RES + m0 + mt * 16 + hg * 8;
            const float4* hp = (const float4*)(hin + base);
            float4 p0 = hp[0], p1 = hp[1];
            float4 o0, o1;
            o0.x = p0.x + acc2[mt][nt][0] + brv[0];
            o0.y = p0.y + acc2[mt][nt][1] + brv[1];
            o0.z = p0.z + acc2[mt][nt][2] + brv[2];
            o0.w = p0.w + acc2[mt][nt][3] + brv[3];
            o1.x = p1.x + acc2[mt][nt][4] + brv[4];
            o1.y = p1.y + acc2[mt][nt][5] + brv[5];
            o1.z = p1.z + acc2[mt][nt][6] + brv[6];
            o1.w = p1.w + acc2[mt][nt][7] + brv[7];
            float4* op = (float4*)(hout + base);
            op[0] = o0; op[1] = o1;
        }
    }
}

// =====================================================================
// Head: skip_sum(t=T-1) from per-layer gated columns, then relu->pw->relu->pw.
// Exact f32; tiny (17.3M MACs total).
// =====================================================================
__global__ void head_kernel(const float* __restrict__ Glast,
                            const float* __restrict__ Ws, const float* __restrict__ bs,
                            const float* __restrict__ We1, const float* __restrict__ be1,
                            const float* __restrict__ We2, const float* __restrict__ be2,
                            float* __restrict__ out) {
    __shared__ float sk[SKIP];
    __shared__ float e1[SKIP];
    int b = blockIdx.x, s = threadIdx.x;
    float acc = 0.f;
    for (int i = 0; i < NL; ++i) {
        const float* g = Glast + ((size_t)i * BATCH + b) * RES;
        const float* w = Ws + ((size_t)i * SKIP + s) * RES;
        float a = 0.f;
        for (int c = 0; c < RES; ++c) a += w[c] * g[c];
        acc += a + bs[(size_t)i * SKIP + s];
    }
    sk[s] = fmaxf(acc, 0.f);
    __syncthreads();
    float a2 = be1[s];
    const float* w1 = We1 + (size_t)s * SKIP;
    for (int c = 0; c < SKIP; ++c) a2 += w1[c] * sk[c];
    e1[s] = fmaxf(a2, 0.f);
    __syncthreads();
    if (s < NCLS) {
        float a3 = be2[s];
        const float* w2 = We2 + (size_t)s * SKIP;
        for (int c = 0; c < SKIP; ++c) a3 += w2[c] * e1[c];
        out[(size_t)b * NCLS + s] = a3;
    }
}

// =====================================================================
extern "C" void kernel_launch(void* const* d_in, const int* in_sizes, int n_in,
                              void* d_out, int out_size, void* d_ws, size_t ws_size,
                              hipStream_t stream) {
    (void)in_sizes; (void)n_in; (void)out_size;
    const float* x      = (const float*)d_in[0];
    const float* Wstart = (const float*)d_in[1];
    const float* bstart = (const float*)d_in[2];
    const float* Wd     = (const float*)d_in[3];
    const float* bd     = (const float*)d_in[4];
    const float* Wr     = (const float*)d_in[5];
    const float* br     = (const float*)d_in[6];
    const float* Ws     = (const float*)d_in[7];
    const float* bs     = (const float*)d_in[8];
    const float* We1    = (const float*)d_in[9];
    const float* be1    = (const float*)d_in[10];
    const float* We2    = (const float*)d_in[11];
    const float* be2    = (const float*)d_in[12];
    float* out = (float*)d_out;

    char* p = (char*)d_ws;
    auto carve = [&](size_t bytes) { char* r = p; p += (bytes + 255) & ~(size_t)255; return r; };
    float*    hA    = (float*)   carve((size_t)BATCH * WWIN * RES * 4);   // 16.8 MB
    float*    hB    = (float*)   carve((size_t)BATCH * WWIN * RES * 4);   // 16.8 MB
    _Float16* WdP   = (_Float16*)carve((size_t)NL * 16 * 16 * 512 * 2);   // 7.9 MB
    _Float16* WrP   = (_Float16*)carve((size_t)NL * 8  * 16 * 512 * 2);   // 3.9 MB
    float*    Glast = (float*)   carve((size_t)NL * BATCH * RES * 4);     // 123 KB
    if ((size_t)(p - (char*)d_ws) > ws_size) return;   // insufficient scratch

    (void)hipFuncSetAttribute((const void*)wavenet_layer,
                              hipFuncAttributeMaxDynamicSharedMemorySize, SMEM_BYTES);

    repack_wd<<<(NL * 16 * 16 * 512 + 255) / 256, 256, 0, stream>>>(Wd, WdP);
    repack_wr<<<(NL * 8  * 16 * 512 + 255) / 256, 256, 0, stream>>>(Wr, WrP);
    start_conv<<<dim3(WWIN, BATCH), 256, 0, stream>>>(x, Wstart, bstart, hA);

    // Per-layer computed tile counts (from the back): nt[i-1] = nt[i] + ceil(d_i/TN).
    int nt[NL];
    nt[NL - 1] = 1;
    for (int i = NL - 1; i >= 1; --i) {
        int d = 1 << (i % 10);
        int v = nt[i] + (d + TN - 1) / TN;
        nt[i - 1] = v > (WWIN / TN) ? (WWIN / TN) : v;
    }

    float* hin = hA; float* hout = hB;
    for (int i = 0; i < NL; ++i) {
        int d = 1 << (i % 10);
        wavenet_layer<<<dim3(nt[i], BATCH), 256, SMEM_BYTES, stream>>>(
            hin, hout,
            WdP + (size_t)i * 16 * 16 * 512,
            WrP + (size_t)i * 8 * 16 * 512,
            bd + (size_t)i * RES, br + (size_t)i * RES,
            Glast + (size_t)i * BATCH * RES, d);
        float* tmp = hin; hin = hout; hout = tmp;
    }

    head_kernel<<<BATCH, SKIP, 0, stream>>>(Glast, Ws, bs, We1, be1, We2, be2, out);
}
